// EnsembleEnvProb_88957362635136
// MI455X (gfx1250) — compile-verified, matrix operand order checked by reading, not run on hardware
//
#include <hip/hip_runtime.h>
#include <hip/hip_bf16.h>

typedef __attribute__((ext_vector_type(16))) _Float16 v16h;
typedef __attribute__((ext_vector_type(8)))  _Float16 v8h;
typedef __attribute__((ext_vector_type(8)))  float    v8f;

#define ENS   7
#define SDIM  30
#define ADIM  8
#define HID   200
#define NHID  3
#define KIN   38      // SDIM + ADIM
#define KINP  64      // padded input K (2 chunks of 32)
#define HIDP  224     // padded hidden (14 tiles of 16, 7 chunks of 32)
#define OUTD  31
#define NT_H  14      // HIDP/16
#define KC_IN 2       // KINP/32
#define KC_H  7       // HIDP/32
#define NT_O  2       // 32 output cols padded
#define ROWS  64      // batch rows per workgroup
#define RT    4       // row tiles of 16

#define W1P_ELEMS  ((long)ENS * NT_H * KC_IN * 512)
#define WHP_ELEMS  ((long)NHID * ENS * NT_H * KC_H * 512)
#define WOP_ELEMS  ((long)ENS * NT_O * KC_H * 512)

// ---------------------------------------------------------------------------
// Pack f32 weights [nMats][K][N] row-major into f16 WMMA B-fragments.
// Fragment layout per (mat, colTile j, kChunk kc): 32 lanes x 16 halfs,
// lane<16:  n=lane,    K = kc*32 + {0..7, 16..23}
// lane>=16: n=lane-16, K = kc*32 + {8..15, 24..31}
// Out-of-range (k>=K || n>=N) packed as zero -> padding is inert, which also
// lets the hidden-layer epilogues skip per-lane column guards.
// ---------------------------------------------------------------------------
__global__ void pack_w(const float* __restrict__ src, _Float16* __restrict__ dst,
                       int nMats, int K, int N, int NT, int KC) {
  long t = (long)blockIdx.x * blockDim.x + threadIdx.x;
  long total = (long)nMats * NT * KC * 512;
  if (t >= total) return;
  int i    = (int)(t & 15);
  int lane = (int)((t >> 4) & 31);
  long rest = t >> 9;
  int kc  = (int)(rest % KC); rest /= KC;
  int j   = (int)(rest % NT);
  int mat = (int)(rest / NT);
  int n  = (lane & 15) + 16 * j;
  int hi = lane >> 4;
  int k  = kc * 32 + ((i < 8) ? (i + 8 * hi) : (16 + (i - 8) + 8 * hi));
  float v = 0.0f;
  if (k < K && n < N) v = src[((long)mat * K + k) * N + n];
  dst[t] = (_Float16)v;
}

// A fragment: 16x32 f16 tile of row-major LDS activations (row stride = HIDP)
__device__ __forceinline__ v16h load_a_lds(const _Float16* buf, int lane, int kc) {
  int row = lane & 15;
  int hi  = lane >> 4;
  const _Float16* p = buf + (long)row * HIDP + kc * 32 + 8 * hi;
  union { v16h v; v8h h[2]; } u;
  u.h[0] = *(const v8h*)(p);       // K = kc*32 + 8*hi + {0..7}
  u.h[1] = *(const v8h*)(p + 16);  // K = kc*32 + 16 + 8*hi + {0..7}
  return u.v;
}

// B fragment: one contiguous 32B load per lane from the packed weights
__device__ __forceinline__ v16h load_b_pk(const _Float16* p32, int lane) {
  const _Float16* p = p32 + lane * 16;
  union { v16h v; v8h h[2]; } u;
  u.h[0] = *(const v8h*)(p);
  u.h[1] = *(const v8h*)(p + 8);
  return u.v;
}

__device__ __forceinline__ float swish_f(float x) {
  // x * sigmoid(x); single v_rcp_f32 instead of IEEE div sequence
  return x * __builtin_amdgcn_rcpf(1.0f + __expf(-x));
}
__device__ __forceinline__ float softplus_f(float x) {
  return fmaxf(x, 0.0f) + __logf(1.0f + __expf(-fabsf(x)));
}

// ---------------------------------------------------------------------------
// One workgroup = (64-row batch tile, ensemble e). 14 waves, one 16-col tile
// of the hidden layer per wave; each wave runs 4 row-tiles against one shared
// B fragment (4x weight reuse). Activations ping-pong in LDS as f16.
// Padded columns (200..223) carry finite junk that is annihilated by the
// zero-padded K rows of the next layer's packed weights -> no guards needed.
// ---------------------------------------------------------------------------
__global__ __launch_bounds__(448) void ensemble_mlp(
    const float* __restrict__ s, const float* __restrict__ a,
    const float* __restrict__ b1, const float* __restrict__ bh,
    const float* __restrict__ bmu, const float* __restrict__ bsig,
    const float* __restrict__ max_s, const float* __restrict__ min_s,
    const float* __restrict__ max_r, const float* __restrict__ min_r,
    const _Float16* __restrict__ W1p, const _Float16* __restrict__ Whp,
    const _Float16* __restrict__ Wmup, const _Float16* __restrict__ Wsigp,
    float* __restrict__ out, int N) {

  __shared__ __align__(16) _Float16 hbuf[2][ROWS][HIDP];

  const int tid  = threadIdx.x;
  const int wave = tid >> 5;
  const int lane = tid & 31;
  const int e    = blockIdx.y;
  const int row0 = blockIdx.x * ROWS;

  // ---- stage 0: load + f16-convert concat(s,a) tile into hbuf[0][:, 0:64)
  for (int idx = tid; idx < ROWS * KINP; idx += 448) {
    int r = idx / KINP, c = idx % KINP;
    int n = row0 + r;
    float v = 0.0f;
    if (n < N) {
      if (c < SDIM)     v = s[((long)n * ENS + e) * SDIM + c];
      else if (c < KIN) v = a[((long)n * ENS + e) * ADIM + (c - SDIM)];
    }
    hbuf[0][r][c] = (_Float16)v;
  }
  __syncthreads();

  const int j    = wave;                   // hidden col tile 0..13
  const int col  = j * 16 + (lane & 15);   // output column of this lane's C frag
  const int hi   = lane >> 4;
  const int cokl = (col < HID) ? col : 0;  // clamped (unconditional bias load)

  // ---- layer 1: 38(->64) x 200(->224)
  {
    v8f acc[RT] = {};
    const _Float16* wb = W1p + (long)(e * NT_H + j) * KC_IN * 512;
    #pragma unroll
    for (int kc = 0; kc < KC_IN; ++kc) {
      v16h bf = load_b_pk(wb + kc * 512, lane);
      #pragma unroll
      for (int rt = 0; rt < RT; ++rt) {
        v16h af = load_a_lds(&hbuf[0][rt * 16][0], lane, kc);
        acc[rt] = __builtin_amdgcn_wmma_f32_16x16x32_f16(false, af, false, bf,
                                                         (short)0, acc[rt], false, false);
      }
    }
    float bias = b1[e * HID + cokl];
    #pragma unroll
    for (int rt = 0; rt < RT; ++rt) {
      #pragma unroll
      for (int v = 0; v < 8; ++v) {
        int r = rt * 16 + v + 8 * hi;
        hbuf[1][r][col] = (_Float16)swish_f(acc[rt][v] + bias);
      }
    }
  }
  __syncthreads();

  // ---- hidden layers: 3 x (200->200), padded to 224x224
  int cur = 1;
  for (int L = 0; L < NHID; ++L) {
    v8f acc[RT] = {};
    const _Float16* wb = Whp + ((long)(L * ENS + e) * NT_H + j) * KC_H * 512;
    #pragma unroll
    for (int kc = 0; kc < KC_H; ++kc) {
      v16h bf = load_b_pk(wb + kc * 512, lane);
      #pragma unroll
      for (int rt = 0; rt < RT; ++rt) {
        v16h af = load_a_lds(&hbuf[cur][rt * 16][0], lane, kc);
        acc[rt] = __builtin_amdgcn_wmma_f32_16x16x32_f16(false, af, false, bf,
                                                         (short)0, acc[rt], false, false);
      }
    }
    float bias = bh[(long)(L * ENS + e) * HID + cokl];
    #pragma unroll
    for (int rt = 0; rt < RT; ++rt) {
      #pragma unroll
      for (int v = 0; v < 8; ++v) {
        int r = rt * 16 + v + 8 * hi;
        hbuf[cur ^ 1][r][col] = (_Float16)swish_f(acc[rt][v] + bias);
      }
    }
    __syncthreads();
    cur ^= 1;
  }

  // ---- heads: waves 0,1 -> mu tiles 0,1 ; waves 2,3 -> sigma tiles 0,1
  if (wave < 4) {
    const int  sel = wave >> 1;       // 0 = mu, 1 = sigma
    const int  jo  = wave & 1;        // output col tile
    const _Float16* Wp = (sel == 0) ? Wmup : Wsigp;
    const float*    bp = (sel == 0) ? bmu  : bsig;

    v8f acc[RT] = {};
    const _Float16* wb = Wp + (long)(e * NT_O + jo) * KC_H * 512;
    #pragma unroll
    for (int kc = 0; kc < KC_H; ++kc) {
      v16h bf = load_b_pk(wb + kc * 512, lane);
      #pragma unroll
      for (int rt = 0; rt < RT; ++rt) {
        v16h af = load_a_lds(&hbuf[cur][rt * 16][0], lane, kc);
        acc[rt] = __builtin_amdgcn_wmma_f32_16x16x32_f16(false, af, false, bf,
                                                         (short)0, acc[rt], false, false);
      }
    }

    const int oc = jo * 16 + (lane & 15);     // 0..31 (31 = pad)
    if (oc < OUTD) {
      const float bias = bp[e * OUTD + oc];
      const long DS_SZ   = (long)N * ENS * SDIM;
      const long DSG_OFF = DS_SZ;
      const long RMU_OFF = 2 * DS_SZ;
      const long RSG_OFF = RMU_OFF + (long)N * ENS;
      const float mx = (oc < SDIM) ? max_s[oc] : max_r[oc - SDIM];
      const float mn = (oc < SDIM) ? min_s[oc] : min_r[oc - SDIM];
      #pragma unroll
      for (int rt = 0; rt < RT; ++rt) {
        #pragma unroll
        for (int v = 0; v < 8; ++v) {
          int n = row0 + rt * 16 + v + 8 * hi;
          if (n < N) {
            float x = acc[rt][v] + bias;
            if (sel == 0) {                      // mu passthrough
              if (oc < SDIM) out[((long)n * ENS + e) * SDIM + oc] = x;
              else           out[RMU_OFF + (long)n * ENS + e]     = x;
            } else {                             // sigma clamp chain + exp
              float t = x;
              t = (mx - softplus_f(mx - 2.0f * t)) * 0.5f;
              t = (mn + softplus_f(2.0f * t - mn)) * 0.5f;
              t = __expf(t);
              if (oc < SDIM) out[DSG_OFF + ((long)n * ENS + e) * SDIM + oc] = t;
              else           out[RSG_OFF + (long)n * ENS + e]              = t;
            }
          }
        }
      }
    }
  }
}

extern "C" void kernel_launch(void* const* d_in, const int* in_sizes, int n_in,
                              void* d_out, int out_size, void* d_ws, size_t ws_size,
                              hipStream_t stream) {
  const float* s     = (const float*)d_in[0];
  const float* a     = (const float*)d_in[1];
  const float* W1    = (const float*)d_in[2];
  const float* b1    = (const float*)d_in[3];
  const float* Wh    = (const float*)d_in[4];
  const float* bh    = (const float*)d_in[5];
  const float* Wmu   = (const float*)d_in[6];
  const float* bmu   = (const float*)d_in[7];
  const float* Wsig  = (const float*)d_in[8];
  const float* bsig  = (const float*)d_in[9];
  const float* max_s = (const float*)d_in[10];
  const float* min_s = (const float*)d_in[11];
  const float* max_r = (const float*)d_in[12];
  const float* min_r = (const float*)d_in[13];

  const int N = in_sizes[0] / (ENS * SDIM);

  _Float16* W1p   = (_Float16*)d_ws;
  _Float16* Whp   = W1p  + W1P_ELEMS;
  _Float16* Wmup  = Whp  + WHP_ELEMS;
  _Float16* Wsigp = Wmup + WOP_ELEMS;

  // Re-pack weights every call (deterministic; tiny vs. main kernel).
  {
    long t1 = W1P_ELEMS;
    pack_w<<<(int)((t1 + 255) / 256), 256, 0, stream>>>(W1, W1p, ENS, KIN, HID, NT_H, KC_IN);
    long th = WHP_ELEMS;
    pack_w<<<(int)((th + 255) / 256), 256, 0, stream>>>(Wh, Whp, NHID * ENS, HID, HID, NT_H, KC_H);
    long to = WOP_ELEMS;
    pack_w<<<(int)((to + 255) / 256), 256, 0, stream>>>(Wmu, Wmup, ENS, HID, OUTD, NT_O, KC_H);
    pack_w<<<(int)((to + 255) / 256), 256, 0, stream>>>(Wsig, Wsigp, ENS, HID, OUTD, NT_O, KC_H);
  }

  dim3 grid((N + ROWS - 1) / ROWS, ENS);
  ensemble_mlp<<<grid, 448, 0, stream>>>(s, a, b1, bh, bmu, bsig,
                                         max_s, min_s, max_r, min_r,
                                         W1p, Whp, Wmup, Wsigp,
                                         (float*)d_out, N);
}
